// GRASSMerge_83554293776903
// MI455X (gfx1250) — matrix operand order repaired and checked by reading, not compiled, not run on hardware
//
#include <hip/hip_runtime.h>
#include <hip/hip_bf16.h>

#define FEATURE_LENGTH 512
#define EMB 128
#define NUM_EMB 2048
#define BATCH 8192
#define NROWS (BATCH * FEATURE_LENGTH / EMB) // 32768

typedef __bf16 v16bf __attribute__((ext_vector_type(16)));
typedef float  v8f   __attribute__((ext_vector_type(8)));

__device__ __forceinline__ unsigned short bf16_rne_bits(float x) {
    unsigned u = __builtin_bit_cast(unsigned, x);
    unsigned r = u + 0x7FFFu + ((u >> 16) & 1u);
    return (unsigned short)(r >> 16);
}
__device__ __forceinline__ __bf16 bf_hi(float x) {
    return __builtin_bit_cast(__bf16, bf16_rne_bits(x));
}
__device__ __forceinline__ __bf16 bf_lo(float x) {
    unsigned short hs = bf16_rne_bits(x);
    float hf = __builtin_bit_cast(float, ((unsigned)hs) << 16);
    return __builtin_bit_cast(__bf16, bf16_rne_bits(x - hf));
}

// ---- Prep 1: wsq[k] = ||W_k||^2 ----------------------------------------
__global__ __launch_bounds__(256) void vq_prep_wsq(const float* __restrict__ W,
                                                   float* __restrict__ wsq) {
    int n = blockIdx.x * blockDim.x + threadIdx.x; // 0..2047
    const float4* wr = reinterpret_cast<const float4*>(W + (size_t)n * EMB);
    float s = 0.f;
#pragma unroll 8
    for (int i = 0; i < EMB / 4; ++i) {
        float4 v = wr[i];
        s += v.x * v.x + v.y * v.y + v.z * v.z + v.w * v.w;
    }
    wsq[n] = s;
}

// ---- Prep 2: split W into bf16 hi/lo, laid out as B-operand tiles ------
// tile index: ((kb*128 + c*32 + l)*16 + s) holds split(W[kb*16+(l&15)][c*32+K(l>>4,s)])
// with K(h,s) = s<8 ? 8h+s : 16+8h+(s-8); lane l reading 16 contiguous bf16
// is the B-matrix register image (lane%16 = N, elements = K in table order).
__global__ __launch_bounds__(256) void vq_prep_pack(const float* __restrict__ W,
                                                    __bf16* __restrict__ whi,
                                                    __bf16* __restrict__ wlo) {
    int t = blockIdx.x * blockDim.x + threadIdx.x; // 0..16383
    int kb = t >> 7, k = t & 127;
    v16bf vh, vl;
#pragma unroll
    for (int n = 0; n < 16; ++n) {
        float x = W[(size_t)(kb * 16 + (k & 15)) * EMB + (k & ~31) + ((n < 8) ? (8 * ((k >> 4) & 1) + n) : (16 + 8 * ((k >> 4) & 1) + n - 8))];
        vh[n] = bf_hi(x);
        vl[n] = bf_lo(x);
    }
    *reinterpret_cast<v16bf*>(whi + (size_t)t * 16) = vh;
    *reinterpret_cast<v16bf*>(wlo + (size_t)t * 16) = vl;
}

// ---- Main: scores via split-bf16 WMMA, argmin, loss, gather ------------
__global__ __launch_bounds__(256) void vq_main(const float* __restrict__ f,
                                               const float* __restrict__ W,
                                               const float* __restrict__ wsq,
                                               const __bf16* __restrict__ whi,
                                               const __bf16* __restrict__ wlo,
                                               float* __restrict__ loss_out,
                                               float* __restrict__ out) {
    __shared__ float lds_d2[8 * 16];
    __shared__ int   lds_j[8 * 16];

    const int lane    = threadIdx.x & 31;
    const int wid     = threadIdx.x >> 5;
    const int rowbase = blockIdx.x * 128 + wid * 16;
    const int myrow   = rowbase + (lane & 15);
    const int khalf   = lane >> 4;

    // Build A hi/lo registers for 16 rows x 128 dims (ISA 16-bit A layout:
    // lane = M (twice), VGPR v<4 -> K=2v(+khalf*8), v>=4 -> K=16+2(v-4)(+khalf*8))
    v16bf a_hi[4], a_lo[4];
    float psum = 0.f;
    const float* frow = f + (size_t)myrow * EMB;
#pragma unroll
    for (int c = 0; c < 4; ++c) {
#pragma unroll
        for (int v = 0; v < 8; ++v) {
            int kp = c * 32 + ((v < 4) ? (2 * v + khalf * 8)
                                       : (16 + 2 * (v - 4) + khalf * 8));
            float2 x = *reinterpret_cast<const float2*>(frow + kp);
            psum += x.x * x.x + x.y * x.y;
            a_hi[c][2 * v]     = bf_hi(x.x);
            a_lo[c][2 * v]     = bf_lo(x.x);
            a_hi[c][2 * v + 1] = bf_hi(x.y);
            a_lo[c][2 * v + 1] = bf_lo(x.y);
        }
    }
    // full ||f_row||^2: lanes l and l+16 each hold half of row l&15
    float fsq_lane = psum + __shfl_xor(psum, 16, 32);

    float best[8];
    int   bkb[8];
#pragma unroll
    for (int r = 0; r < 8; ++r) { best[r] = -3.4e38f; bkb[r] = 0; }

    const float* wsq_me = wsq + (lane & 15);

    for (int kb = 0; kb < NUM_EMB / 16; ++kb) {
        const __bf16* th = whi + (size_t)kb * (128 * 16);
        const __bf16* tl = wlo + (size_t)kb * (128 * 16);
        v8f acc0 = {0.f, 0.f, 0.f, 0.f, 0.f, 0.f, 0.f, 0.f};
        v8f acc1 = {0.f, 0.f, 0.f, 0.f, 0.f, 0.f, 0.f, 0.f};
#pragma unroll
        for (int c = 0; c < 4; ++c) {
            v16bf bh = *reinterpret_cast<const v16bf*>(th + (size_t)(c * 32 + lane) * 16);
            v16bf bl = *reinterpret_cast<const v16bf*>(tl + (size_t)(c * 32 + lane) * 16);
            acc0 = __builtin_amdgcn_wmma_f32_16x16x32_bf16(false, a_hi[c], false, bh,
                                                           (short)0, acc0, false, false);
            acc1 = __builtin_amdgcn_wmma_f32_16x16x32_bf16(false, a_hi[c], false, bl,
                                                           (short)0, acc1, false, false);
            acc1 = __builtin_amdgcn_wmma_f32_16x16x32_bf16(false, a_lo[c], false, bh,
                                                           (short)0, acc1, false, false);
        }
        // q = s - 0.5*||W||^2 ; argmax q == argmin d2. C layout: lane%16 = code n.
        float halfw = 0.5f * wsq_me[kb * 16];
#pragma unroll
        for (int r = 0; r < 8; ++r) {
            float q  = acc0[r] + acc1[r] - halfw;
            bool  gt = q > best[r];
            best[r] = gt ? q : best[r];
            bkb[r]  = gt ? kb : bkb[r];
        }
    }

    // Cross-lane argmax within 16-lane groups (C layout: VGPR r -> rows r / r+8)
#pragma unroll
    for (int r = 0; r < 8; ++r) {
        float v = best[r];
        int   j = bkb[r] * 16 + (lane & 15);
#pragma unroll
        for (int off = 1; off < 16; off <<= 1) {
            float ov = __shfl_xor(v, off, 32);
            int   oj = __shfl_xor(j, off, 32);
            if (ov > v || (ov == v && oj < j)) { v = ov; j = oj; }
        }
        int   src   = (lane < 16) ? r : (r + 24);  // fsq of row r / row r+8
        float fsq_r = __shfl(fsq_lane, src, 32);
        float d2    = fsq_r - 2.0f * v;
        if (lane == 0)  { lds_d2[wid * 16 + r]     = d2; lds_j[wid * 16 + r]     = j; }
        if (lane == 16) { lds_d2[wid * 16 + r + 8] = d2; lds_j[wid * 16 + r + 8] = j; }
    }
    __syncthreads();

    // loss[b] = (1.25/4) * sum of 4 consecutive sub-row d2min
    if (lane < 4) {
        int base = wid * 16 + lane * 4;
        float s = lds_d2[base] + lds_d2[base + 1] + lds_d2[base + 2] + lds_d2[base + 3];
        loss_out[blockIdx.x * 32 + wid * 4 + lane] = 0.3125f * s;
    }

    // out[row] = W[j[row]] ; 32 lanes x float4 = 128 floats per row
    int j_l = lds_j[wid * 16 + (lane & 15)];
#pragma unroll
    for (int m = 0; m < 16; ++m) {
        int jm = __shfl(j_l, m, 32);
        float4 wv = *reinterpret_cast<const float4*>(W + (size_t)jm * EMB + lane * 4);
        *reinterpret_cast<float4*>(out + (size_t)(rowbase + m) * EMB + lane * 4) = wv;
    }
}

extern "C" void kernel_launch(void* const* d_in, const int* in_sizes, int n_in,
                              void* d_out, int out_size, void* d_ws, size_t ws_size,
                              hipStream_t stream) {
    const float* f = (const float*)d_in[0];            // [8192, 512]
    const float* W = (const float*)d_in[1];            // [2048, 128]
    float* loss_out = (float*)d_out;                   // [8192]
    float* out      = (float*)d_out + BATCH;           // [8192*512]

    float*  wsq = (float*)d_ws;                                        // 8 KB
    __bf16* whi = (__bf16*)((char*)d_ws + 8192);                       // 512 KB
    __bf16* wlo = (__bf16*)((char*)d_ws + 8192 + (size_t)NUM_EMB * EMB * 2);

    vq_prep_wsq<<<NUM_EMB / 256, 256, 0, stream>>>(W, wsq);
    vq_prep_pack<<<(128 * 128) / 256, 256, 0, stream>>>(W, whi, wlo);
    vq_main<<<NROWS / 128, 256, 0, stream>>>(f, W, wsq, whi, wlo, loss_out, out);
}
